// GAT_node_classification_54245436949039
// MI455X (gfx1250) — matrix operand order repaired and checked by reading, not compiled, not run on hardware
//
#include <hip/hip_runtime.h>
#include <hip/hip_bf16.h>

// ---------------------------------------------------------------------------
// CDNA5 (gfx1250) GAT node classification
//   - Dense feature transforms via v_wmma_f32_16x16x32_bf16 (wave32 WMMA)
//     with 32x64 wave tiles (8 WMMAs per K-step, B fragments reused 2x)
//   - Edge softmax/aggregation via VALU + native f32 atomics
// ---------------------------------------------------------------------------

typedef __attribute__((ext_vector_type(16))) __bf16         v16bf;
typedef __attribute__((ext_vector_type(8)))  float          v8f;
typedef __attribute__((ext_vector_type(8)))  unsigned short v8u;
typedef __attribute__((ext_vector_type(16))) unsigned short v16u;

__device__ __forceinline__ unsigned short f32_to_bf16_rne(float f) {
    unsigned int u = __float_as_uint(f);
    unsigned int r = 0x7FFFu + ((u >> 16) & 1u);
    return (unsigned short)((u + r) >> 16);
}

// Monotonic-int trick: float atomic max (memory must be initialized to -inf).
__device__ __forceinline__ void atomicMaxF(float* addr, float val) {
    int bits = __float_as_int(val);
    if (bits >= 0) {
        atomicMax((int*)addr, bits);
    } else {
        atomicMin((unsigned int*)addr, (unsigned int)bits);
    }
}

// ---------------------------------------------------------------------------
// Conversions
// ---------------------------------------------------------------------------
__global__ void cvt_f32_to_bf16(const float* __restrict__ x,
                                unsigned short* __restrict__ y, int n) {
    int i = blockIdx.x * blockDim.x + threadIdx.x;
    if (i < n) y[i] = f32_to_bf16_rne(x[i]);
}

// W [K,Nout] row-major f32  ->  Wt [Nout,K] bf16 (column-contiguous in K)
__global__ void transpose_cvt_bf16(const float* __restrict__ W,
                                   unsigned short* __restrict__ Wt,
                                   int K, int Nout) {
    int i = blockIdx.x * blockDim.x + threadIdx.x;
    if (i >= K * Nout) return;
    int n = i / K;
    int k = i - n * K;
    Wt[i] = f32_to_bf16_rne(W[(size_t)k * Nout + n]);
}

// relu(out1) -> bf16 input for layer 2
__global__ void relu_cvt_bf16(const float* __restrict__ x,
                              unsigned short* __restrict__ y, int n) {
    int i = blockIdx.x * blockDim.x + threadIdx.x;
    if (i < n) {
        float v = x[i];
        y[i] = f32_to_bf16_rne(v > 0.f ? v : 0.f);
    }
}

// ---------------------------------------------------------------------------
// WMMA GEMM:  C[M,Nn] = A[M,K](bf16) x B[K,Nn], with B given transposed as
// Bt[Nn,K](bf16).  Block = 128 threads = 4 waves; block tile 128x64; wave
// tile 32(M) x 64(N): 2 A fragments x 4 B fragments = 8 WMMAs per K-step.
// OOB rows are clamped on load (never zeroed) and masked on store only.
// Fragment layouts per CDNA5 ISA 7.12.2.
// ---------------------------------------------------------------------------
__global__ __launch_bounds__(128)
void gemm_bf16_wmma(const unsigned short* __restrict__ A,
                    const unsigned short* __restrict__ Bt,
                    float* __restrict__ C,
                    int M, int Nn, int K) {
    const int lane  = threadIdx.x & 31;
    const int wave  = threadIdx.x >> 5;
    const int half  = lane >> 4;     // K-half selector
    const int l16   = lane & 15;     // row (A) / col (B,C) within tile
    const int mBase = blockIdx.x * 128 + wave * 32;
    const int nBase = blockIdx.y * 64;

    // Clamp OOB A rows to a valid row; garbage rows are never stored.
    int r0 = mBase + l16;        if (r0 >= M) r0 = M - 1;
    int r1 = mBase + 16 + l16;   if (r1 >= M) r1 = M - 1;
    const size_t aRow0 = (size_t)r0 * (size_t)K;
    const size_t aRow1 = (size_t)r1 * (size_t)K;

    v8f acc[4][2] = {};

    for (int k = 0; k < K; k += 32) {
        // ---- A fragments: 16x32 bf16, lane half -> K octets {h*8, 16+h*8}
        v16u au0, au1;
        {
            const unsigned short* pa0 = A + aRow0 + k + half * 8;
            const unsigned short* pa1 = A + aRow1 + k + half * 8;
            v8u lo0 = *(const v8u*)(pa0);
            v8u hi0 = *(const v8u*)(pa0 + 16);
            v8u lo1 = *(const v8u*)(pa1);
            v8u hi1 = *(const v8u*)(pa1 + 16);
#pragma unroll
            for (int i = 0; i < 8; ++i) {
                au0[i] = lo0[i]; au0[8 + i] = hi0[i];
                au1[i] = lo1[i]; au1[8 + i] = hi1[i];
            }
        }
        v16bf a0 = __builtin_bit_cast(v16bf, au0);
        v16bf a1 = __builtin_bit_cast(v16bf, au1);

#pragma unroll
        for (int t = 0; t < 4; ++t) {
            // ---- B fragment: 32x16 bf16; lane col = l16, lane half -> K
            // octets {h*8, 16+h*8} from the transposed weight row.
            const unsigned short* pb =
                Bt + (size_t)(nBase + t * 16 + l16) * (size_t)K + k + half * 8;
            v16u bu;
            v8u blo = *(const v8u*)(pb);
            v8u bhi = *(const v8u*)(pb + 16);
#pragma unroll
            for (int i = 0; i < 8; ++i) { bu[i] = blo[i]; bu[8 + i] = bhi[i]; }
            v16bf bfrag = __builtin_bit_cast(v16bf, bu);

            acc[t][0] = __builtin_amdgcn_wmma_f32_16x16x32_bf16(
                false, a0, false, bfrag, (short)0, acc[t][0], false, false);
            acc[t][1] = __builtin_amdgcn_wmma_f32_16x16x32_bf16(
                false, a1, false, bfrag, (short)0, acc[t][1], false, false);
        }
    }

    // ---- Store D: VGPR r -> row r + 8*half, col = l16 within 16x16 tile
#pragma unroll
    for (int t = 0; t < 4; ++t) {
#pragma unroll
        for (int m2 = 0; m2 < 2; ++m2) {
#pragma unroll
            for (int r = 0; r < 8; ++r) {
                int row = mBase + m2 * 16 + half * 8 + r;
                if (row < M)
                    C[(size_t)row * Nn + nBase + t * 16 + l16] = acc[t][m2][r];
            }
        }
    }
}

// ---------------------------------------------------------------------------
// Attention scores: one wave per (node, head); two dot products of length C.
// ---------------------------------------------------------------------------
__global__ void node_scores(const float* __restrict__ h,
                            const float* __restrict__ att_src,
                            const float* __restrict__ att_dst,
                            float* __restrict__ a_s,
                            float* __restrict__ a_d,
                            int N, int C) {
    int gt   = blockIdx.x * blockDim.x + threadIdx.x;
    int wid  = gt >> 5;
    int lane = gt & 31;
    int n    = wid >> 2;        // H = 4
    int hd   = wid & 3;
    if (n >= N) return;

    const float* hp = h + (size_t)n * (4 * C) + (size_t)hd * C;
    const float* as = att_src + (size_t)hd * C;
    const float* ad = att_dst + (size_t)hd * C;
    float s1 = 0.f, s2 = 0.f;
    for (int c = lane; c < C; c += 32) {
        float v = hp[c];
        s1 += v * as[c];
        s2 += v * ad[c];
    }
#pragma unroll
    for (int off = 16; off > 0; off >>= 1) {
        s1 += __shfl_xor(s1, off);
        s2 += __shfl_xor(s2, off);
    }
    if (lane == 0) {
        a_s[(size_t)n * 4 + hd] = s1;
        a_d[(size_t)n * 4 + hd] = s2;
    }
}

// ---------------------------------------------------------------------------
// Init kernels
// ---------------------------------------------------------------------------
__global__ void init_attn_state(float* __restrict__ m,
                                float* __restrict__ den, int n) {
    int i = blockIdx.x * blockDim.x + threadIdx.x;
    if (i < n) {
        m[i]   = __uint_as_float(0xFF800000u);   // -inf
        den[i] = 0.f;
    }
}

__global__ void init_out_bias(float* __restrict__ out,
                              const float* __restrict__ bias,
                              int total, int HC) {
    int i = blockIdx.x * blockDim.x + threadIdx.x;
    if (i < total) out[i] = bias[i % HC];
}

// ---------------------------------------------------------------------------
// Edge passes. Edges e < E come from edge_index; e >= E are self loops.
// ---------------------------------------------------------------------------
__global__ void edge_max_pass(const int* __restrict__ ei, int E, int ET,
                              const float* __restrict__ a_s,
                              const float* __restrict__ a_d,
                              float* __restrict__ esc,
                              float* __restrict__ m) {
    int e = blockIdx.x * blockDim.x + threadIdx.x;
    if (e >= ET) return;
    int s, d;
    if (e < E) { s = ei[e]; d = ei[E + e]; } else { s = e - E; d = s; }
    float4 sv = *(const float4*)(a_s + (size_t)s * 4);
    float4 dv = *(const float4*)(a_d + (size_t)d * 4);
    float vals[4] = { sv.x + dv.x, sv.y + dv.y, sv.z + dv.z, sv.w + dv.w };
#pragma unroll
    for (int hd = 0; hd < 4; ++hd) {
        float v = vals[hd];
        v = v > 0.f ? v : 0.2f * v;            // leaky-relu, att slope
        esc[(size_t)e * 4 + hd] = v;
        atomicMaxF(&m[(size_t)d * 4 + hd], v);
    }
}

__global__ void edge_expsum_pass(const int* __restrict__ ei, int E, int ET,
                                 float* __restrict__ esc,
                                 const float* __restrict__ m,
                                 float* __restrict__ den) {
    int e = blockIdx.x * blockDim.x + threadIdx.x;
    if (e >= ET) return;
    int d;
    if (e < E) { d = ei[E + e]; } else { d = e - E; }
    float4 ev = *(const float4*)(esc + (size_t)e * 4);
    float4 mv = *(const float4*)(m + (size_t)d * 4);
    float p0 = __expf(ev.x - mv.x);
    float p1 = __expf(ev.y - mv.y);
    float p2 = __expf(ev.z - mv.z);
    float p3 = __expf(ev.w - mv.w);
    float4 pv = { p0, p1, p2, p3 };
    *(float4*)(esc + (size_t)e * 4) = pv;      // overwrite score with p
    atomicAdd(&den[(size_t)d * 4 + 0], p0);
    atomicAdd(&den[(size_t)d * 4 + 1], p1);
    atomicAdd(&den[(size_t)d * 4 + 2], p2);
    atomicAdd(&den[(size_t)d * 4 + 3], p3);
}

// One thread per (edge, 4-channel chunk): float4 load + 4 atomic f32 adds.
__global__ void edge_aggregate_pass(const int* __restrict__ ei, int E, int ET,
                                    const float* __restrict__ hfeat,
                                    const float* __restrict__ esc,
                                    const float* __restrict__ den,
                                    float* __restrict__ out,
                                    int C, int HC) {
    const int nCh = HC >> 2;
    long long flat = (long long)blockIdx.x * blockDim.x + threadIdx.x;
    long long total = (long long)ET * nCh;
    if (flat >= total) return;
    int e  = (int)(flat / nCh);
    int ch = ((int)(flat - (long long)e * nCh)) << 2;
    int s, d;
    if (e < E) { s = ei[e]; d = ei[E + e]; } else { s = e - E; d = s; }
    int hd = ch / C;
    float alpha = esc[(size_t)e * 4 + hd] /
                  (den[(size_t)d * 4 + hd] + 1e-16f);
    float4 v = *(const float4*)(hfeat + (size_t)s * HC + ch);
    float* o = out + (size_t)d * HC + ch;
    atomicAdd(o + 0, v.x * alpha);
    atomicAdd(o + 1, v.y * alpha);
    atomicAdd(o + 2, v.z * alpha);
    atomicAdd(o + 3, v.w * alpha);
}

// ---------------------------------------------------------------------------
// Final: leaky_relu(0.01) then [N,128] @ Wl[128,7] + bl
// ---------------------------------------------------------------------------
__global__ void final_linear(const float* __restrict__ h,
                             const float* __restrict__ Wl,
                             const float* __restrict__ bl,
                             float* __restrict__ out, int N) {
    int n = blockIdx.x * blockDim.x + threadIdx.x;
    if (n >= N) return;
    float acc[7];
#pragma unroll
    for (int c = 0; c < 7; ++c) acc[c] = bl[c];
    const float* hp = h + (size_t)n * 128;
    for (int i = 0; i < 128; ++i) {
        float v = hp[i];
        v = v > 0.f ? v : 0.01f * v;
#pragma unroll
        for (int c = 0; c < 7; ++c) acc[c] += v * Wl[i * 7 + c];
    }
#pragma unroll
    for (int c = 0; c < 7; ++c) out[(size_t)n * 7 + c] = acc[c];
}

// ---------------------------------------------------------------------------
// Host launch
// ---------------------------------------------------------------------------
extern "C" void kernel_launch(void* const* d_in, const int* in_sizes, int n_in,
                              void* d_out, int out_size, void* d_ws, size_t ws_size,
                              hipStream_t stream) {
    const float* x        = (const float*)d_in[0];
    const int*   ei       = (const int*)d_in[1];     // [2,E] src row then dst row
    const float* W1       = (const float*)d_in[2];
    const float* att_src1 = (const float*)d_in[3];
    const float* att_dst1 = (const float*)d_in[4];
    const float* b1       = (const float*)d_in[5];
    const float* W2       = (const float*)d_in[6];
    const float* att_src2 = (const float*)d_in[7];
    const float* att_dst2 = (const float*)d_in[8];
    const float* b2       = (const float*)d_in[9];
    const float* Wl       = (const float*)d_in[10];
    const float* bl       = (const float*)d_in[11];
    float* y = (float*)d_out;

    const int Fin = 256, HC1 = 256, HC2 = 128, H = 4, C1 = 64, C2 = 32;
    const int N  = in_sizes[0] / Fin;
    const int E  = in_sizes[1] / 2;
    const int ET = E + N;                // with self loops

    // Workspace carve-up (256B aligned)
    char* ws = (char*)d_ws;
    size_t off = 0;
    auto carve = [&](size_t bytes) -> void* {
        void* p = ws + off;
        off += (bytes + 255) & ~(size_t)255;
        return p;
    };
    unsigned short* xb   = (unsigned short*)carve((size_t)N * Fin * 2);   // also h2 f32 later
    unsigned short* w1t  = (unsigned short*)carve((size_t)HC1 * Fin * 2);
    unsigned short* w2t  = (unsigned short*)carve((size_t)HC2 * HC1 * 2);
    float*          h1   = (float*)carve((size_t)N * HC1 * 4);            // also out2 later
    float*          a_s  = (float*)carve((size_t)N * H * 4);
    float*          a_d  = (float*)carve((size_t)N * H * 4);
    float*          mbuf = (float*)carve((size_t)N * H * 4);
    float*          den  = (float*)carve((size_t)N * H * 4);
    float*          esc  = (float*)carve((size_t)ET * H * 4);
    float*          out1 = (float*)carve((size_t)N * HC1 * 4);
    unsigned short* h1b  = (unsigned short*)carve((size_t)N * HC1 * 2);
    float*          h2   = (float*)xb;    // reuse: N*HC2*4 == N*Fin*2 bytes
    float*          out2 = h1;            // reuse: h1 dead after layer-1 aggregation

    const int TB = 256;
    auto blocks = [](long long n, int tb) { return (unsigned)((n + tb - 1) / tb); };

    // ---- Layer 1 ----
    cvt_f32_to_bf16<<<blocks((long long)N * Fin, TB), TB, 0, stream>>>(x, xb, N * Fin);
    transpose_cvt_bf16<<<blocks((long long)Fin * HC1, TB), TB, 0, stream>>>(W1, w1t, Fin, HC1);
    transpose_cvt_bf16<<<blocks((long long)HC1 * HC2, TB), TB, 0, stream>>>(W2, w2t, HC1, HC2);

    {
        dim3 g((N + 127) / 128, HC1 / 64);
        gemm_bf16_wmma<<<g, 128, 0, stream>>>(xb, w1t, h1, N, HC1, Fin);
    }
    node_scores<<<blocks((long long)N * H * 32, TB), TB, 0, stream>>>(
        h1, att_src1, att_dst1, a_s, a_d, N, C1);
    init_attn_state<<<blocks((long long)N * H, TB), TB, 0, stream>>>(mbuf, den, N * H);
    init_out_bias<<<blocks((long long)N * HC1, TB), TB, 0, stream>>>(out1, b1, N * HC1, HC1);

    edge_max_pass<<<blocks(ET, TB), TB, 0, stream>>>(ei, E, ET, a_s, a_d, esc, mbuf);
    edge_expsum_pass<<<blocks(ET, TB), TB, 0, stream>>>(ei, E, ET, esc, mbuf, den);
    edge_aggregate_pass<<<blocks((long long)ET * (HC1 / 4), TB), TB, 0, stream>>>(
        ei, E, ET, h1, esc, den, out1, C1, HC1);

    relu_cvt_bf16<<<blocks((long long)N * HC1, TB), TB, 0, stream>>>(out1, h1b, N * HC1);

    // ---- Layer 2 ----
    {
        dim3 g((N + 127) / 128, HC2 / 64);
        gemm_bf16_wmma<<<g, 128, 0, stream>>>(h1b, w2t, h2, N, HC2, HC1);
    }
    node_scores<<<blocks((long long)N * H * 32, TB), TB, 0, stream>>>(
        h2, att_src2, att_dst2, a_s, a_d, N, C2);
    init_attn_state<<<blocks((long long)N * H, TB), TB, 0, stream>>>(mbuf, den, N * H);
    init_out_bias<<<blocks((long long)N * HC2, TB), TB, 0, stream>>>(out2, b2, N * HC2, HC2);

    edge_max_pass<<<blocks(ET, TB), TB, 0, stream>>>(ei, E, ET, a_s, a_d, esc, mbuf);
    edge_expsum_pass<<<blocks(ET, TB), TB, 0, stream>>>(ei, E, ET, esc, mbuf, den);
    edge_aggregate_pass<<<blocks((long long)ET * (HC2 / 4), TB), TB, 0, stream>>>(
        ei, E, ET, h2, esc, den, out2, C2, HC2);

    // ---- Classifier ----
    final_linear<<<blocks(N, TB), TB, 0, stream>>>(out2, Wl, bl, y, N);
}
